// V2VModule_26259430048622
// MI455X (gfx1250) — matrix operand order repaired and checked by reading, not compiled, not run on hardware
//
#include <hip/hip_runtime.h>
#include <stdint.h>

#define NV    200000
#define KOFF  27
#define CIN   16
#define CMID  32
#define COUT  32
#define EPSB  1e-3f
#define NTILE (NV / 16)     // 12500 output tiles of 16 rows
#define RPART 256           // partial-sum blocks for BN reduction
#define WBYTES (KOFF * 1024 * 2)   // swizzled weights: 27*2*32*16 bf16 = 55296 B
#define WVEC4  (WBYTES / 16)       // 3456 uint4

typedef __bf16 bf16;
typedef __attribute__((ext_vector_type(16))) __bf16 v16bf;
typedef __attribute__((ext_vector_type(8)))  __bf16 v8bf;
typedef __attribute__((ext_vector_type(8)))  float  v8f;

// ---------------------------------------------------------------- rulebook inversion
__global__ __launch_bounds__(256) void init_inv(int* __restrict__ inv) {
    int i = blockIdx.x * blockDim.x + threadIdx.x;
    if (i < KOFF * NV) inv[i] = NV;   // NV = zero-padded row
}

__global__ __launch_bounds__(256) void scatter_inv(const int* __restrict__ rin,
                                                   const int* __restrict__ rout,
                                                   int* __restrict__ inv) {
    int i = blockIdx.x * blockDim.x + threadIdx.x;
    if (i >= KOFF * NV) return;
    int o = rout[i];
    if (o < NV) {
        int k = i / NV;
        inv[(size_t)k * NV + o] = rin[i];   // injective per k -> no conflicts
    }
}

// ---------------------------------------------------------------- input / weight prep
__global__ __launch_bounds__(256) void prep_feats(const float* __restrict__ feats,
                                                  uint16_t* __restrict__ xb) {
    int i = blockIdx.x * blockDim.x + threadIdx.x;
    if (i >= (NV + 1) * CIN) return;
    float v = (i < NV * CIN) ? feats[i] : 0.f;
    ((bf16*)xb)[i] = (bf16)v;
}

// Swizzle W[k][K][n] (row-major KxN per offset) into the WMMA B-operand layout:
// flat index i = k*1024 + t*512 + L*16 + e ; K = (e&7) + 8*(L>=16) + 16*(e>=8) ; n = (L&15) + 16*t
template <int CHIN>
__global__ __launch_bounds__(256) void prep_w(const float* __restrict__ W,
                                              uint16_t* __restrict__ wsw) {
    int i = blockIdx.x * blockDim.x + threadIdx.x;
    if (i >= KOFF * 1024) return;
    int e = i & 15;
    int L = (i >> 4) & 31;
    int t = (i >> 9) & 1;
    int k = i >> 10;
    int Kd = (e & 7) + 8 * (L >> 4) + 16 * (e >> 3);
    int n  = (L & 15) + 16 * t;
    float v = (Kd < CHIN) ? W[((size_t)k * CHIN + Kd) * 32 + n] : 0.f;
    ((bf16*)wsw)[i] = (bf16)v;
}

// ---------------------------------------------------------------- WMMA conv (gather at output)
// One wave per 16-row output tile.  A: 16 gathered rows x CHIN (K padded to 32),
// B: weight tiles staged once per block into LDS, C/D: two v8f accumulators.
template <int CHIN>
__global__ __launch_bounds__(256) void conv_wmma(const uint16_t* __restrict__ xb_u,
                                                 const uint16_t* __restrict__ wsw_u,
                                                 const int* __restrict__ inv,
                                                 float* __restrict__ y) {
    const bf16* __restrict__ xb = (const bf16*)xb_u;

    __shared__ uint4 smw4[WVEC4];                       // 55296 B of swizzled weights

    const int tid  = threadIdx.x;
    const int lane = tid & 31;
    const int wave = tid >> 5;

    // Cooperative stage of all 27 B tiles into LDS (runs on all threads, pre-guard
    // so the barrier is non-divergent).
    {
        const uint4* g = (const uint4*)wsw_u;
        for (int i = tid; i < WVEC4; i += 256) smw4[i] = g[i];
    }
    __syncthreads();
    const bf16* __restrict__ smw = (const bf16*)smw4;

    const int tile = blockIdx.x * 8 + wave;
    if (tile >= NTILE) return;               // wave-uniform: EXEC stays all-ones for WMMA

    const int base = tile * 16;
    const int rsel = base + (lane & 15);     // output row handled by this lane (A: M = lane%16)
    const int half = lane >> 4;              // lane half selects K sub-range of A

    // Preload all 27 gather indices into registers (enables prefetching ahead).
    int ridx[KOFF];
#pragma unroll
    for (int k = 0; k < KOFF; ++k) ridx[k] = inv[(size_t)k * NV + rsel];

    v8f acc0 = {};
    v8f acc1 = {};

#pragma unroll 3
    for (int k = 0; k < KOFF; ++k) {
        if (k + 2 < KOFF)
            __builtin_prefetch(xb + (size_t)ridx[k + 2] * CHIN, 0, 3);

        const bf16* arow = xb + (size_t)ridx[k] * CHIN;

        // ISA 16-bit A layout: lanes<16 hold K 0-7 / 16-23, lanes>=16 hold K 8-15 / 24-31.
        v16bf a;
        v8bf lo = *(const v8bf*)(arow + half * 8);
        if (CHIN == 16) {
            v8bf zz = {};
            a = __builtin_shufflevector(lo, zz, 0, 1, 2, 3, 4, 5, 6, 7,
                                               8, 9, 10, 11, 12, 13, 14, 15);
        } else {
            v8bf hi8 = *(const v8bf*)(arow + 16 + half * 8);
            a = __builtin_shufflevector(lo, hi8, 0, 1, 2, 3, 4, 5, 6, 7,
                                                8, 9, 10, 11, 12, 13, 14, 15);
        }

        // B operands from LDS (ds_load path; 16B/lane -> conflict-free 2-pass access)
        const v16bf b0 = *(const v16bf*)(smw + (((k * 2 + 0) * 32 + lane) * 16));
        const v16bf b1 = *(const v16bf*)(smw + (((k * 2 + 1) * 32 + lane) * 16));

        acc0 = __builtin_amdgcn_wmma_f32_16x16x32_bf16(false, a, false, b0,
                                                       (short)0, acc0, false, false);
        acc1 = __builtin_amdgcn_wmma_f32_16x16x32_bf16(false, a, false, b1,
                                                       (short)0, acc1, false, false);
    }

    // C/D layout: VGPR r, lane L -> M = r + 8*(L>=16), N = L&15.
    const int col = lane & 15;
#pragma unroll
    for (int r = 0; r < 8; ++r) {
        int row = base + r + 8 * half;
        y[(size_t)row * 32 + col]      = acc0[r];
        y[(size_t)row * 32 + 16 + col] = acc1[r];
    }
}

// ---------------------------------------------------------------- BN statistics (deterministic 2-phase)
__global__ __launch_bounds__(256) void reduce_stats(const float* __restrict__ y,
                                                    float* __restrict__ part) {
    __shared__ float ss[256];
    __shared__ float sq[256];
    int tid = threadIdx.x;
    int col = tid & 31;
    int rl  = tid >> 5;
    float s = 0.f, q = 0.f;
    for (int row = blockIdx.x * 8 + rl; row < NV; row += RPART * 8) {
        float v = y[(size_t)row * 32 + col];
        s += v;
        q += v * v;
    }
    ss[tid] = s;
    sq[tid] = q;
    __syncthreads();
    for (int off = 128; off >= 32; off >>= 1) {
        if (tid < off) { ss[tid] += ss[tid + off]; sq[tid] += sq[tid + off]; }
        __syncthreads();
    }
    if (tid < 32) {
        part[blockIdx.x * 64 + tid]      = ss[tid];
        part[blockIdx.x * 64 + 32 + tid] = sq[tid];
    }
}

__global__ void finalize_stats(const float* __restrict__ part,
                               const float* __restrict__ gamma,
                               const float* __restrict__ beta,
                               float* __restrict__ stats) {
    __shared__ float tot[64];
    int t = threadIdx.x;   // 64 threads
    float a = 0.f;
    for (int p = 0; p < RPART; ++p) a += part[p * 64 + t];
    tot[t] = a;
    __syncthreads();
    if (t < 32) {
        float mean = tot[t] / (float)NV;
        float var  = tot[32 + t] / (float)NV - mean * mean;   // biased var (training BN)
        float sc   = gamma[t] * rsqrtf(var + EPSB);
        stats[t]      = sc;
        stats[32 + t] = beta[t] - mean * sc;
    }
}

// ---------------------------------------------------------------- BN apply + ReLU
__global__ __launch_bounds__(256) void bn_relu_bf16(const float* __restrict__ y,
                                                    const float* __restrict__ stats,
                                                    uint16_t* __restrict__ xout) {
    int i = blockIdx.x * blockDim.x + threadIdx.x;
    if (i >= (NV + 1) * 32) return;
    if (i < NV * 32) {
        int c = i & 31;
        float o = fmaxf(fmaf(y[i], stats[c], stats[32 + c]), 0.f);
        ((bf16*)xout)[i] = (bf16)o;
    } else {
        ((bf16*)xout)[i] = (bf16)0.f;   // zero-pad row NV
    }
}

__global__ __launch_bounds__(256) void bn_relu_f32(const float* __restrict__ y,
                                                   const float* __restrict__ stats,
                                                   float* __restrict__ out) {
    int i = blockIdx.x * blockDim.x + threadIdx.x;
    if (i >= NV * 32) return;
    int c = i & 31;
    out[i] = fmaxf(fmaf(y[i], stats[c], stats[32 + c]), 0.f);
}

// ---------------------------------------------------------------- host launch
extern "C" void kernel_launch(void* const* d_in, const int* in_sizes, int n_in,
                              void* d_out, int out_size, void* d_ws, size_t ws_size,
                              hipStream_t stream) {
    const float* feats = (const float*)d_in[0];
    const float* W1    = (const float*)d_in[1];
    const float* g1    = (const float*)d_in[2];
    const float* b1    = (const float*)d_in[3];
    const float* W2    = (const float*)d_in[4];
    const float* g2    = (const float*)d_in[5];
    const float* b2    = (const float*)d_in[6];
    const int*   rin   = (const int*)d_in[7];
    const int*   rout  = (const int*)d_in[8];
    float*       out   = (float*)d_out;

    char* p = (char*)d_ws;
    auto carve = [&](size_t bytes) -> void* {
        void* q = (void*)p;
        p += (bytes + 255) & ~(size_t)255;
        return q;
    };
    int*      inv   = (int*)      carve((size_t)KOFF * NV * sizeof(int));       // 21.6 MB
    uint16_t* xb0   = (uint16_t*) carve((size_t)(NV + 1) * CIN * 2);            //  6.4 MB
    uint16_t* x1b   = (uint16_t*) carve((size_t)(NV + 1) * CMID * 2);           // 12.8 MB
    uint16_t* w1s   = (uint16_t*) carve((size_t)WBYTES);                        // 55 KB
    uint16_t* w2s   = (uint16_t*) carve((size_t)WBYTES);                        // 55 KB
    float*    y     = (float*)    carve((size_t)NV * 32 * sizeof(float));       // 25.6 MB
    float*    part  = (float*)    carve((size_t)RPART * 64 * sizeof(float));
    float*    stats = (float*)    carve(64 * sizeof(float));

    const int tinv = KOFF * NV;
    init_inv   <<<(tinv + 255) / 256, 256, 0, stream>>>(inv);
    scatter_inv<<<(tinv + 255) / 256, 256, 0, stream>>>(rin, rout, inv);
    prep_feats <<<((NV + 1) * CIN + 255) / 256, 256, 0, stream>>>(feats, xb0);
    prep_w<CIN> <<<(KOFF * 1024 + 255) / 256, 256, 0, stream>>>(W1, w1s);
    prep_w<CMID><<<(KOFF * 1024 + 255) / 256, 256, 0, stream>>>(W2, w2s);

    // layer 1: conv -> stats -> BN+ReLU -> bf16 activations
    conv_wmma<CIN><<<(NTILE + 7) / 8, 256, 0, stream>>>(xb0, w1s, inv, y);
    reduce_stats  <<<RPART, 256, 0, stream>>>(y, part);
    finalize_stats<<<1, 64, 0, stream>>>(part, g1, b1, stats);
    bn_relu_bf16  <<<((NV + 1) * 32 + 255) / 256, 256, 0, stream>>>(y, stats, x1b);

    // layer 2: conv -> stats -> BN+ReLU -> f32 output
    conv_wmma<CMID><<<(NTILE + 7) / 8, 256, 0, stream>>>(x1b, w2s, inv, y);
    reduce_stats  <<<RPART, 256, 0, stream>>>(y, part);
    finalize_stats<<<1, 64, 0, stream>>>(part, g2, b2, stats);
    bn_relu_f32   <<<(NV * 32 + 255) / 256, 256, 0, stream>>>(y, stats, out);
}
